// DynamicGroup_21517786153538
// MI455X (gfx1250) — matrix operand order repaired and checked by reading, not compiled, not run on hardware
//
#include <hip/hip_runtime.h>
#include <hip/hip_bf16.h>

// Problem constants (match reference)
#define BB 128
#define SS 2048
#define EE 512
#define HH 512

typedef __bf16 v16bf __attribute__((ext_vector_type(16)));
typedef __bf16 v8bf  __attribute__((ext_vector_type(8)));
typedef float  v8f   __attribute__((ext_vector_type(8)));

__device__ __forceinline__ unsigned short f2bf(float f) {
    unsigned u = __float_as_uint(f);
    u += 0x7FFFu + ((u >> 16) & 1u);          // round-to-nearest-even
    return (unsigned short)(u >> 16);
}

// ---------------------------------------------------------------------------
// Ua_w (f32, [H][H] row-major: o x k) -> bf16 copy in workspace
__global__ __launch_bounds__(256)
void k_ua_convert(const float* __restrict__ ua, unsigned short* __restrict__ ua16) {
    int i0 = (blockIdx.x * 256 + threadIdx.x) * 4;   // 256*256*4 = 262144 = H*H
    float4 f = *(const float4*)(ua + i0);
    ua16[i0 + 0] = f2bf(f.x);
    ua16[i0 + 1] = f2bf(f.y);
    ua16[i0 + 2] = f2bf(f.z);
    ua16[i0 + 3] = f2bf(f.w);
}

// ---------------------------------------------------------------------------
// qq[b][o] = Wa_b[o] + Ua_b[o] + sum_h h_prev[b][h] * Wa_w[o][h]
__global__ __launch_bounds__(256)
void k_qq(const float* __restrict__ hp, const float* __restrict__ Waw,
          const float* __restrict__ Wab, const float* __restrict__ Uab,
          float* __restrict__ qq) {
    __shared__ float h[HH];
    int b = blockIdx.x;
    for (int i = threadIdx.x; i < HH; i += 256) h[i] = hp[b * HH + i];
    __syncthreads();
    for (int o = threadIdx.x; o < HH; o += 256) {
        float s = Wab[o] + Uab[o];
        const float4* w = (const float4*)(Waw + (size_t)o * HH);
        #pragma unroll 4
        for (int j = 0; j < HH / 4; ++j) {
            float4 f = w[j];
            s += f.x * h[4*j] + f.y * h[4*j+1] + f.z * h[4*j+2] + f.w * h[4*j+3];
        }
        qq[b * HH + o] = s;
    }
}

// ---------------------------------------------------------------------------
// Fused scores kernel. One block = one batch b, one 64-row s-tile.
// scores[b][s] = sum_o tanh( qq[b][o] + (x_ref[b,s,:] . Ua_w[o,:]) ) * Va[o]
// (Va_b dropped: constant shift is softmax-invariant.)
__global__ __launch_bounds__(256)
void k_scores(const float* __restrict__ xref,          // [B][S][H] f32
              const unsigned short* __restrict__ uab,  // [H(o)][H(k)] bf16
              const float* __restrict__ qq,            // [B][H]
              const float* __restrict__ va,            // [H]
              float* __restrict__ scores)              // [B][S]
{
    __shared__ unsigned short Atile[64 * HH];   // 64 KB, bf16 x tile
    __shared__ float wred[8][64];

    int wg = blockIdx.x;               // 0..4095
    int b  = wg >> 5;                  // S/64 = 32 tiles per batch
    int st = wg & 31;
    int s0 = st * 64;
    int tid = threadIdx.x;

    // Load + convert A tile: 64 rows x 512 k, f32 -> bf16 (row-major in LDS)
    const float* src = xref + ((size_t)b * SS + s0) * HH;
    for (int i = tid; i < 64 * HH / 4; i += 256) {
        float4 f = ((const float4*)src)[i];
        int base = i * 4;
        Atile[base + 0] = f2bf(f.x);
        Atile[base + 1] = f2bf(f.y);
        Atile[base + 2] = f2bf(f.z);
        Atile[base + 3] = f2bf(f.w);
    }
    __syncthreads();

    int lane  = tid & 31;
    int wave  = tid >> 5;      // 0..7 : each wave owns 4 o-tiles (64 features)
    int lhalf = lane >> 4;     // 0|1
    int l16   = lane & 15;

    union ABu { v16bf v; v8bf h[2]; };

    float acc[4][8] = {};
    #pragma unroll
    for (int nt4 = 0; nt4 < 4; ++nt4) {
        int nt = wave * 4 + nt4;
        int n  = nt * 16 + l16;                       // this lane's output col
        const unsigned short* brow = uab + (size_t)n * HH;

        v8f c0 = {}, c1 = {}, c2 = {}, c3 = {};
        for (int k = 0; k < HH; k += 32) {
            // B fragment: lane holds col n, K = k+16*lhalf .. +15 (contiguous)
            v16bf bf = *reinterpret_cast<const v16bf*>(brow + k + 16 * lhalf);
            #pragma unroll
            for (int mt = 0; mt < 4; ++mt) {
                const unsigned short* arow = Atile + (mt * 16 + l16) * HH;
                ABu u;
                u.h[0] = *reinterpret_cast<const v8bf*>(arow + k + 16 * lhalf);
                u.h[1] = *reinterpret_cast<const v8bf*>(arow + k + 16 * lhalf + 8);
                v8f* cp = (mt == 0) ? &c0 : (mt == 1) ? &c1 : (mt == 2) ? &c2 : &c3;
                *cp = __builtin_amdgcn_wmma_f32_16x16x32_bf16(
                          false, u.v, false, bf, (short)0, *cp, false, false);
            }
        }
        float qn = qq[b * HH + n];
        float vn = va[n];
        v8f cc[4] = {c0, c1, c2, c3};
        #pragma unroll
        for (int mt = 0; mt < 4; ++mt)
            #pragma unroll
            for (int r = 0; r < 8; ++r)
                acc[mt][r] += tanhf(cc[mt][r] + qn) * vn;
    }

    // Deterministic reduction: lanes within each 16-lane half hold N-partials
    // of the same row (M = mt*16 + r + 8*lhalf).
    #pragma unroll
    for (int mt = 0; mt < 4; ++mt)
        #pragma unroll
        for (int r = 0; r < 8; ++r) {
            float v = acc[mt][r];
            v += __shfl_xor(v, 1, 32);
            v += __shfl_xor(v, 2, 32);
            v += __shfl_xor(v, 4, 32);
            v += __shfl_xor(v, 8, 32);
            if (l16 == 0) wred[wave][mt * 16 + r + 8 * lhalf] = v;
        }
    __syncthreads();
    if (tid < 64) {
        float s = 0.f;
        #pragma unroll
        for (int w = 0; w < 8; ++w) s += wred[w][tid];
        scores[(size_t)b * SS + s0 + tid] = s;
    }
}

// ---------------------------------------------------------------------------
// In-place softmax over S per batch row.
__global__ __launch_bounds__(256)
void k_softmax(float* __restrict__ sc) {
    __shared__ float red[256];
    int b = blockIdx.x, tid = threadIdx.x;
    float v[8];
    float mx = -1e30f;
    #pragma unroll
    for (int i = 0; i < 8; ++i) {
        v[i] = sc[(size_t)b * SS + tid + i * 256];
        mx = fmaxf(mx, v[i]);
    }
    red[tid] = mx; __syncthreads();
    for (int off = 128; off > 0; off >>= 1) {
        if (tid < off) red[tid] = fmaxf(red[tid], red[tid + off]);
        __syncthreads();
    }
    mx = red[0]; __syncthreads();
    float sum = 0.f;
    #pragma unroll
    for (int i = 0; i < 8; ++i) { v[i] = __expf(v[i] - mx); sum += v[i]; }
    red[tid] = sum; __syncthreads();
    for (int off = 128; off > 0; off >>= 1) {
        if (tid < off) red[tid] += red[tid + off];
        __syncthreads();
    }
    float inv = 1.0f / red[0];
    #pragma unroll
    for (int i = 0; i < 8; ++i) sc[(size_t)b * SS + tid + i * 256] = v[i] * inv;
}

// ---------------------------------------------------------------------------
// Context partials: block (b, c) accumulates 256 s-values into part[b][c][h].
__global__ __launch_bounds__(256)
void k_context(const float* __restrict__ attn, const float* __restrict__ xref,
               float* __restrict__ part) {
    __shared__ float w[256];
    int b = blockIdx.x, c = blockIdx.y;
    int s0 = c * 256;
    w[threadIdx.x] = attn[(size_t)b * SS + s0 + threadIdx.x];
    __syncthreads();
    int h0 = threadIdx.x, h1 = threadIdx.x + 256;
    float a0 = 0.f, a1 = 0.f;
    const float* xp = xref + ((size_t)b * SS + s0) * HH;
    for (int s = 0; s < 256; ++s) {
        float ws = w[s];
        a0 += ws * xp[(size_t)s * HH + h0];
        a1 += ws * xp[(size_t)s * HH + h1];
    }
    float* pp = part + ((size_t)b * 8 + c) * HH;
    pp[h0] = a0; pp[h1] = a1;
}

// ---------------------------------------------------------------------------
// RNN cell. Reduces the 8 context partials (fixed order -> deterministic),
// writes context to d_out's second half, h_next to the first half.
__global__ __launch_bounds__(256)
void k_rnn(const float* __restrict__ xt, const float* __restrict__ part,
           const float* __restrict__ hp,
           const float* __restrict__ Wih, const float* __restrict__ Wihb,
           const float* __restrict__ Whh, const float* __restrict__ Whhb,
           float* __restrict__ out) {
    __shared__ float xin[EE + HH];
    __shared__ float hh[HH];
    int b = blockIdx.x, tid = threadIdx.x;
    for (int i = tid; i < HH; i += 256) {
        float cv = 0.f;
        #pragma unroll
        for (int c = 0; c < 8; ++c) cv += part[((size_t)b * 8 + c) * HH + i];
        xin[i]      = xt[(size_t)b * EE + i];
        xin[EE + i] = cv;
        hh[i]       = hp[(size_t)b * HH + i];
        out[BB * HH + (size_t)b * HH + i] = cv;    // context output
    }
    __syncthreads();
    for (int o = tid; o < HH; o += 256) {
        float s = Wihb[o] + Whhb[o];
        const float4* wi = (const float4*)(Wih + (size_t)o * (EE + HH));
        #pragma unroll 4
        for (int j = 0; j < (EE + HH) / 4; ++j) {
            float4 f = wi[j];
            s += f.x * xin[4*j] + f.y * xin[4*j+1] + f.z * xin[4*j+2] + f.w * xin[4*j+3];
        }
        const float4* wh = (const float4*)(Whh + (size_t)o * HH);
        #pragma unroll 4
        for (int j = 0; j < HH / 4; ++j) {
            float4 f = wh[j];
            s += f.x * hh[4*j] + f.y * hh[4*j+1] + f.z * hh[4*j+2] + f.w * hh[4*j+3];
        }
        out[(size_t)b * HH + o] = tanhf(s);
    }
}

// ---------------------------------------------------------------------------
extern "C" void kernel_launch(void* const* d_in, const int* in_sizes, int n_in,
                              void* d_out, int out_size, void* d_ws, size_t ws_size,
                              hipStream_t stream) {
    const float* x_t    = (const float*)d_in[0];
    const float* xref   = (const float*)d_in[1];
    const float* h_prev = (const float*)d_in[2];
    const float* Wa_w   = (const float*)d_in[3];
    const float* Wa_b   = (const float*)d_in[4];
    const float* Ua_w   = (const float*)d_in[5];
    const float* Ua_b   = (const float*)d_in[6];
    const float* Va_w   = (const float*)d_in[7];
    // d_in[8] = Va_b : constant shift before softmax, mathematically dropped
    const float* Wih_w  = (const float*)d_in[9];
    const float* Wih_b  = (const float*)d_in[10];
    const float* Whh_w  = (const float*)d_in[11];
    const float* Whh_b  = (const float*)d_in[12];
    float* out = (float*)d_out;

    char* ws = (char*)d_ws;
    float*          qq     = (float*)(ws);                         // 256 KB
    unsigned short* ua16   = (unsigned short*)(ws + 262144);       // 512 KB
    float*          scores = (float*)(ws + 786432);                // 1 MB
    float*          part   = (float*)(ws + 1835008);               // 2 MB
    // total workspace footprint: 3.75 MB

    k_ua_convert<<<256, 256, 0, stream>>>(Ua_w, ua16);
    k_qq<<<BB, 256, 0, stream>>>(h_prev, Wa_w, Wa_b, Ua_b, qq);
    k_scores<<<BB * (SS / 64), 256, 0, stream>>>(xref, ua16, qq, Va_w, scores);
    k_softmax<<<BB, 256, 0, stream>>>(scores);
    k_context<<<dim3(BB, 8), 256, 0, stream>>>(scores, xref, part);
    k_rnn<<<BB, 256, 0, stream>>>(x_t, part, h_prev, Wih_w, Wih_b, Whh_w, Whh_b, out);
}
